// Attention_13125420056644
// MI455X (gfx1250) — compile-verified, matrix operand order checked by reading, not run on hardware
//
#include <hip/hip_runtime.h>
#include <hip/hip_fp16.h>

// CDNA5 (gfx1250) wave32 WMMA attention pipeline.
// All matmuls via v_wmma_f32_16x16x32_f16 (f16 in, f32 accumulate).

typedef __attribute__((ext_vector_type(16))) _Float16 v16h;
typedef __attribute__((ext_vector_type(8)))  _Float16 v8h;
typedef __attribute__((ext_vector_type(4)))  _Float16 v4h;
typedef __attribute__((ext_vector_type(8)))  float    v8f;

#define LOG2E 1.44269504088896340736f

// Load a 16-wide f16 A/B fragment (16x32 tile, this lane's row).
// CDNA5 16-bit A layout: lanes 0-15 hold K={0..7,16..23}, lanes 16-31 hold
// K={8..15,24..31}; v16h elements 0..7 -> VGPR0-3, 8..15 -> VGPR4-7.
static __device__ __forceinline__ v16h load_frag16(const _Float16* __restrict__ p, int hiOff) {
  v8h c0 = *(const v8h*)(p + hiOff);        // K = hiOff .. hiOff+7
  v8h c1 = *(const v8h*)(p + 16 + hiOff);   // K = 16+hiOff .. 23+hiOff
  v16h r;
#pragma unroll
  for (int i = 0; i < 8; ++i) { r[i] = c0[i]; r[8 + i] = c1[i]; }
  return r;
}

static __device__ __forceinline__ v8f wmma_f16(v16h a, v16h b, v8f c) {
  return __builtin_amdgcn_wmma_f32_16x16x32_f16(false, a, false, b, (short)0, c, false, false);
}

// ---------------------------------------------------------------- RMSNorm
// One wave per 512-element row; 16 f32 per lane; wave shfl reduction.
__global__ void k_rmsnorm(const float* __restrict__ x, const float* __restrict__ g,
                          _Float16* __restrict__ xn) {
  int row  = blockIdx.x * 8 + (threadIdx.x >> 5);
  int lane = threadIdx.x & 31;
  const float4* p = (const float4*)(x + (size_t)row * 512);
  float4 v[4];
  float ssq = 0.f;
#pragma unroll
  for (int i = 0; i < 4; ++i) {
    v[i] = p[lane + 32 * i];
    ssq += v[i].x * v[i].x + v[i].y * v[i].y + v[i].z * v[i].z + v[i].w * v[i].w;
  }
#pragma unroll
  for (int m = 1; m < 32; m <<= 1) ssq += __shfl_xor(ssq, m, 32);
  float r = rsqrtf(ssq * (1.f / 512.f) + 1.1920929e-7f);  // eps = FLT_EPSILON
  const float4* gp = (const float4*)g;
#pragma unroll
  for (int i = 0; i < 4; ++i) {
    float4 gw = gp[lane + 32 * i];
    v4h o;
    o[0] = (_Float16)(v[i].x * r * gw.x);
    o[1] = (_Float16)(v[i].y * r * gw.y);
    o[2] = (_Float16)(v[i].z * r * gw.z);
    o[3] = (_Float16)(v[i].w * r * gw.w);
    *(v4h*)(xn + (size_t)row * 512 + (size_t)(lane + 32 * i) * 4) = o;
  }
}

// ------------------------------------------- weight transpose + f32->f16
__global__ void k_transpose_f16(const float* __restrict__ w, _Float16* __restrict__ wt,
                                int K, int N) {
  int idx = blockIdx.x * blockDim.x + threadIdx.x;
  if (idx >= K * N) return;
  int k = idx / N, n = idx - k * N;
  wt[(size_t)n * K + k] = (_Float16)w[idx];
}

// Scatter one 16x16 f32 C tile into Q/K/Vt f16 layouts.
static __device__ __forceinline__ void store_qkv_tile(v8f c, int m0, int j, int hiOff,
                                                      _Float16* __restrict__ Qm,
                                                      _Float16* __restrict__ Km,
                                                      _Float16* __restrict__ Vt) {
  int s = j >> 9;              // 0=q 1=k 2=v
  int h = (j >> 6) & 7;
  int d = j & 63;
#pragma unroll
  for (int i = 0; i < 8; ++i) {
    int m = m0 + i + hiOff;    // C layout: vgpr i -> row i (+8 for hi half)
    int b = m >> 12;
    int n = m & 4095;
    _Float16 val = (_Float16)c[i];
    size_t bh = (size_t)(b * 8 + h);
    if (s == 0)      Qm[(bh * 4096 + n) * 64 + d] = val;
    else if (s == 1) Km[(bh * 4096 + n) * 64 + d] = val;
    else             Vt[(bh * 64 + d) * 4096 + n] = val;
  }
}

// ------------------------------------------------------------- QKV GEMM
// [8192,512] @ [512,1536]; one wave -> 32x32 C macro-tile (2x2 WMMA tiles,
// each A/B fragment reused twice -> 2 b128 loads per WMMA).
__global__ void k_gemm_qkv(const _Float16* __restrict__ X, const _Float16* __restrict__ WT,
                           _Float16* __restrict__ Qm, _Float16* __restrict__ Km,
                           _Float16* __restrict__ Vt) {
  int lane = threadIdx.x & 31;
  int wave = threadIdx.x >> 5;
  int lrow = lane & 15;
  int hiOff = (lane >= 16) ? 8 : 0;
  int n0 = (blockIdx.x * 8 + wave) * 32;
  int m0 = blockIdx.y * 32;
  const _Float16* ar0 = X  + (size_t)(m0 + lrow) * 512;
  const _Float16* ar1 = X  + (size_t)(m0 + 16 + lrow) * 512;
  const _Float16* br0 = WT + (size_t)(n0 + lrow) * 512;
  const _Float16* br1 = WT + (size_t)(n0 + 16 + lrow) * 512;
  v8f c00 = {}, c01 = {}, c10 = {}, c11 = {};
#pragma unroll 2
  for (int k = 0; k < 512; k += 32) {
    v16h a0 = load_frag16(ar0 + k, hiOff);
    v16h a1 = load_frag16(ar1 + k, hiOff);
    v16h b0 = load_frag16(br0 + k, hiOff);
    v16h b1 = load_frag16(br1 + k, hiOff);
    c00 = wmma_f16(a0, b0, c00);
    c01 = wmma_f16(a0, b1, c01);
    c10 = wmma_f16(a1, b0, c10);
    c11 = wmma_f16(a1, b1, c11);
  }
  store_qkv_tile(c00, m0,      n0 + lrow,      hiOff, Qm, Km, Vt);
  store_qkv_tile(c01, m0,      n0 + 16 + lrow, hiOff, Qm, Km, Vt);
  store_qkv_tile(c10, m0 + 16, n0 + lrow,      hiOff, Qm, Km, Vt);
  store_qkv_tile(c11, m0 + 16, n0 + 16 + lrow, hiOff, Qm, Km, Vt);
}

// --------------------------------------------------------- flash attention
// Per wave: 32 query rows (two 16-row sets A/B), keys streamed in tiles of 32.
// S^T = K.Q^T puts queries in lanes: softmax stats are per-lane + one
// shfl_xor(16), and P^T (C layout) IS the B fragment for O^T = V^T.P^T.
// K/V fragments are each reused by both query sets -> 16 loads : 16 WMMAs.
__global__ void k_flash(const _Float16* __restrict__ Qm, const _Float16* __restrict__ Km,
                        const _Float16* __restrict__ Vt, _Float16* __restrict__ AO) {
  int lane = threadIdx.x & 31;
  int wave = threadIdx.x >> 5;
  int lrow = lane & 15;
  int hiOff = (lane >= 16) ? 8 : 0;
  int bh = blockIdx.y;
  int q0 = (blockIdx.x * 8 + wave) * 32;
  const _Float16* Qb = Qm + (size_t)bh * 4096 * 64;
  const _Float16* Kb = Km + (size_t)bh * 4096 * 64;
  const _Float16* Vb = Vt + (size_t)bh * 64 * 4096;

  // Q as B fragments (lane = query column), dh split 0..31 / 32..63
  const _Float16* qrA = Qb + (size_t)(q0 + lrow) * 64;
  const _Float16* qrB = Qb + (size_t)(q0 + 16 + lrow) * 64;
  v16h bq0a = load_frag16(qrA, hiOff);
  v16h bq1a = load_frag16(qrA + 32, hiOff);
  v16h bq0b = load_frag16(qrB, hiOff);
  v16h bq1b = load_frag16(qrB + 32, hiOff);

  v8f oa0 = {}, oa1 = {}, oa2 = {}, oa3 = {};   // O^T for query set A
  v8f ob0 = {}, ob1 = {}, ob2 = {}, ob3 = {};   // O^T for query set B
  float rmaxA = -1e30f, rsumA = 0.f;
  float rmaxB = -1e30f, rsumB = 0.f;

  for (int j0 = 0; j0 < 4096; j0 += 32) {
    // ---- S^T = K . Q^T : A = K rows (keys), two 16-key fragments, shared
    const _Float16* kr0 = Kb + (size_t)(j0 + lrow) * 64;
    const _Float16* kr1 = Kb + (size_t)(j0 + 16 + lrow) * 64;
    v16h k00 = load_frag16(kr0,      hiOff);
    v16h k01 = load_frag16(kr0 + 32, hiOff);
    v16h k10 = load_frag16(kr1,      hiOff);
    v16h k11 = load_frag16(kr1 + 32, hiOff);
    v8f sA0 = {}, sA1 = {}, sB0 = {}, sB1 = {};
    sA0 = wmma_f16(k00, bq0a, sA0);  sA0 = wmma_f16(k01, bq1a, sA0);
    sA1 = wmma_f16(k10, bq0a, sA1);  sA1 = wmma_f16(k11, bq1a, sA1);
    sB0 = wmma_f16(k00, bq0b, sB0);  sB0 = wmma_f16(k01, bq1b, sB0);
    sB1 = wmma_f16(k10, bq0b, sB1);  sB1 = wmma_f16(k11, bq1b, sB1);

    // ---- online softmax, set A (lane holds keys {i+hiOff},{16+i+hiOff})
    float lmax = -1e30f;
#pragma unroll
    for (int i = 0; i < 8; ++i) lmax = fmaxf(lmax, fmaxf(sA0[i], sA1[i]));
    lmax = fmaxf(lmax, __shfl_xor(lmax, 16, 32));
    float nmaxA  = fmaxf(rmaxA, lmax);
    float scaleA = exp2f((rmaxA - nmaxA) * LOG2E);
    v16h bpA;                     // P^T in C layout == B-fragment layout
    float lsum = 0.f;
#pragma unroll
    for (int i = 0; i < 8; ++i) {
      float p0 = exp2f((sA0[i] - nmaxA) * LOG2E);
      float p1 = exp2f((sA1[i] - nmaxA) * LOG2E);
      lsum += p0 + p1;
      bpA[i]     = (_Float16)p0;
      bpA[8 + i] = (_Float16)p1;
    }
    lsum += __shfl_xor(lsum, 16, 32);
    rsumA = rsumA * scaleA + lsum;
    rmaxA = nmaxA;

    // ---- online softmax, set B
    lmax = -1e30f;
#pragma unroll
    for (int i = 0; i < 8; ++i) lmax = fmaxf(lmax, fmaxf(sB0[i], sB1[i]));
    lmax = fmaxf(lmax, __shfl_xor(lmax, 16, 32));
    float nmaxB  = fmaxf(rmaxB, lmax);
    float scaleB = exp2f((rmaxB - nmaxB) * LOG2E);
    v16h bpB;
    lsum = 0.f;
#pragma unroll
    for (int i = 0; i < 8; ++i) {
      float p0 = exp2f((sB0[i] - nmaxB) * LOG2E);
      float p1 = exp2f((sB1[i] - nmaxB) * LOG2E);
      lsum += p0 + p1;
      bpB[i]     = (_Float16)p0;
      bpB[8 + i] = (_Float16)p1;
    }
    lsum += __shfl_xor(lsum, 16, 32);
    rsumB = rsumB * scaleB + lsum;
    rmaxB = nmaxB;

#pragma unroll
    for (int i = 0; i < 8; ++i) {
      oa0[i] *= scaleA; oa1[i] *= scaleA; oa2[i] *= scaleA; oa3[i] *= scaleA;
      ob0[i] *= scaleB; ob1[i] *= scaleB; ob2[i] *= scaleB; ob3[i] *= scaleB;
    }

    // ---- O^T += V^T . P^T : each V fragment feeds both query sets
    v16h v0 = load_frag16(Vb + (size_t)(lrow)      * 4096 + j0, hiOff);
    v16h v1 = load_frag16(Vb + (size_t)(16 + lrow) * 4096 + j0, hiOff);
    v16h v2 = load_frag16(Vb + (size_t)(32 + lrow) * 4096 + j0, hiOff);
    v16h v3 = load_frag16(Vb + (size_t)(48 + lrow) * 4096 + j0, hiOff);
    oa0 = wmma_f16(v0, bpA, oa0);  ob0 = wmma_f16(v0, bpB, ob0);
    oa1 = wmma_f16(v1, bpA, oa1);  ob1 = wmma_f16(v1, bpB, ob1);
    oa2 = wmma_f16(v2, bpA, oa2);  ob2 = wmma_f16(v2, bpB, ob2);
    oa3 = wmma_f16(v3, bpA, oa3);  ob3 = wmma_f16(v3, bpB, ob3);
  }

  float invA = 1.f / rsumA;
  float invB = 1.f / rsumB;
  int b = bh >> 3, h = bh & 7;
  _Float16* orA = AO + (size_t)(b * 4096 + q0 + lrow) * 512 + h * 64;
  _Float16* orB = AO + (size_t)(b * 4096 + q0 + 16 + lrow) * 512 + h * 64;
#pragma unroll
  for (int i = 0; i < 8; ++i) {
    orA[ 0 + i + hiOff] = (_Float16)(oa0[i] * invA);
    orA[16 + i + hiOff] = (_Float16)(oa1[i] * invA);
    orA[32 + i + hiOff] = (_Float16)(oa2[i] * invA);
    orA[48 + i + hiOff] = (_Float16)(oa3[i] * invA);
    orB[ 0 + i + hiOff] = (_Float16)(ob0[i] * invB);
    orB[16 + i + hiOff] = (_Float16)(ob1[i] * invB);
    orB[32 + i + hiOff] = (_Float16)(ob2[i] * invB);
    orB[48 + i + hiOff] = (_Float16)(ob3[i] * invB);
  }
}

// ----------------------------------------------------------- output GEMM
// [8192,512] @ [512,512]; 32x32 macro-tile per wave, f32 out.
__global__ void k_gemm_out(const _Float16* __restrict__ A, const _Float16* __restrict__ WT,
                           float* __restrict__ out) {
  int lane = threadIdx.x & 31;
  int wave = threadIdx.x >> 5;
  int lrow = lane & 15;
  int hiOff = (lane >= 16) ? 8 : 0;
  int n0 = (blockIdx.x * 8 + wave) * 32;
  int m0 = blockIdx.y * 32;
  const _Float16* ar0 = A  + (size_t)(m0 + lrow) * 512;
  const _Float16* ar1 = A  + (size_t)(m0 + 16 + lrow) * 512;
  const _Float16* br0 = WT + (size_t)(n0 + lrow) * 512;
  const _Float16* br1 = WT + (size_t)(n0 + 16 + lrow) * 512;
  v8f c00 = {}, c01 = {}, c10 = {}, c11 = {};
#pragma unroll 2
  for (int k = 0; k < 512; k += 32) {
    v16h a0 = load_frag16(ar0 + k, hiOff);
    v16h a1 = load_frag16(ar1 + k, hiOff);
    v16h b0 = load_frag16(br0 + k, hiOff);
    v16h b1 = load_frag16(br1 + k, hiOff);
    c00 = wmma_f16(a0, b0, c00);
    c01 = wmma_f16(a0, b1, c01);
    c10 = wmma_f16(a1, b0, c10);
    c11 = wmma_f16(a1, b1, c11);
  }
#pragma unroll
  for (int i = 0; i < 8; ++i) {
    out[(size_t)(m0 + i + hiOff) * 512      + n0 + lrow]      = c00[i];
    out[(size_t)(m0 + i + hiOff) * 512      + n0 + 16 + lrow] = c01[i];
    out[(size_t)(m0 + 16 + i + hiOff) * 512 + n0 + lrow]      = c10[i];
    out[(size_t)(m0 + 16 + i + hiOff) * 512 + n0 + 16 + lrow] = c11[i];
  }
}

extern "C" void kernel_launch(void* const* d_in, const int* in_sizes, int n_in,
                              void* d_out, int out_size, void* d_ws, size_t ws_size,
                              hipStream_t stream) {
  (void)in_sizes; (void)n_in; (void)out_size; (void)ws_size;
  const float* tokens = (const float*)d_in[0];
  const float* gamma  = (const float*)d_in[1];
  const float* w_qkv  = (const float*)d_in[2];   // [512, 1536]
  const float* w_out  = (const float*)d_in[3];   // [512, 512]
  float* out = (float*)d_out;                    // [2, 4096, 512]

  char* ws = (char*)d_ws;
  _Float16* xn    = (_Float16*)(ws +        0);  //  8 MB: [8192, 512]
  _Float16* Qm    = (_Float16*)(ws +  8388608);  //  8 MB: [16, 4096, 64]
  _Float16* Km    = (_Float16*)(ws + 16777216);  //  8 MB: [16, 4096, 64]
  _Float16* Vt    = (_Float16*)(ws + 25165824);  //  8 MB: [16, 64, 4096]
  _Float16* AO    = (_Float16*)(ws + 33554432);  //  8 MB: [8192, 512]
  _Float16* WqkvT = (_Float16*)(ws + 41943040);  // 1.5 MB: [1536, 512]
  _Float16* WoutT = (_Float16*)(ws + 43515904);  // 0.5 MB: [512, 512]

  k_transpose_f16<<<(512 * 1536 + 255) / 256, 256, 0, stream>>>(w_qkv, WqkvT, 512, 1536);
  k_transpose_f16<<<(512 * 512  + 255) / 256, 256, 0, stream>>>(w_out, WoutT, 512, 512);
  k_rmsnorm<<<8192 / 8, 256, 0, stream>>>(tokens, gamma, xn);
  k_gemm_qkv<<<dim3(6, 256), 256, 0, stream>>>(xn, WqkvT, Qm, Km, Vt);
  k_flash<<<dim3(16, 16), 256, 0, stream>>>(Qm, Km, Vt, AO);
  k_gemm_out<<<dim3(2, 256), 256, 0, stream>>>(AO, WoutT, out);
}
